// TriangleAttentionUngated_68178310857259
// MI455X (gfx1250) — matrix operand order repaired and checked by reading, not compiled, not run on hardware
//
#include <hip/hip_runtime.h>
#include <hip/hip_bf16.h>

typedef _Float16 half_t;
typedef __attribute__((ext_vector_type(16))) _Float16 v16h;
typedef __attribute__((ext_vector_type(8)))  float    v8f;

#define NDIM 192
#define RDIM (192*192)        // 36864 rows
#define SCALE 0.25f           // 16^-0.5
#define SSH 210               // f16 score row stride: word stride 105, gcd(105,64)=1 -> conflict free
#define SVT 200               // V-transposed row stride (halves)

// ---------- WMMA fragment helpers (CDNA5 wave32 layouts) ----------
// A (16x32 f16): lane m=lane&15 is row; lane<16 holds K {0..7,16..23}, lane>=16 holds K {8..15,24..31}
__device__ inline v16h frag_a(const half_t* base, int ld, int r0, int kb, int lane) {
  int m = lane & 15, hi = lane >> 4;
  const half_t* p = base + (size_t)(r0 + m) * ld + kb + hi * 8;
  v16h a;
#pragma unroll
  for (int x = 0; x < 8; ++x) a[x] = p[x];
#pragma unroll
  for (int x = 0; x < 8; ++x) a[x + 8] = p[16 + x];
  return a;
}
// B (32x16 f16) sourced from a row-major [N][K] buffer (i.e. pre-transposed operand):
// lane n=lane&15 is column; lane<16 holds K {0..7,16..23}, lane>=16 holds K {8..15,24..31}
__device__ inline v16h frag_b(const half_t* base, int ld, int n0, int kb, int lane) {
  int n = lane & 15, hi = lane >> 4;
  const half_t* p = base + (size_t)(n0 + n) * ld + kb + hi * 8;
  v16h b;
#pragma unroll
  for (int x = 0; x < 8; ++x) b[x] = p[x];
#pragma unroll
  for (int x = 0; x < 8; ++x) b[x + 8] = p[16 + x];
  return b;
}

// ---------- weight prep: transpose + f16 convert ----------
__global__ void k_prep_w(const float* __restrict__ Wq0, const float* __restrict__ Wq1,
                         const float* __restrict__ Wo,
                         half_t* __restrict__ Wt0, half_t* __restrict__ Wt1,
                         half_t* __restrict__ WoT) {
  int idx = blockIdx.x * 256 + threadIdx.x;
  if (idx < 2 * 384 * 128) {
    int p = idx / (384 * 128);
    int rem = idx % (384 * 128);
    int c = rem / 128, k = rem % 128;
    const float* W = p ? Wq1 : Wq0;
    half_t* Wt = p ? Wt1 : Wt0;
    Wt[c * 128 + k] = (half_t)W[k * 384 + c];          // Wt[c][k] = W[k][c]
  } else {
    int rem = idx - 2 * 384 * 128;
    if (rem < 128 * 256) {
      int c = rem / 256, k = rem % 256;
      WoT[c * 256 + k] = (half_t)Wo[k * 128 + c];      // WoT[c][k] = Wo[k][c]
    }
  }
}

// ---------- LayerNorm -> f16 : one wave32 per 128-wide row ----------
__global__ void k_ln(const float* __restrict__ e, const float* __restrict__ g,
                     const float* __restrict__ b, half_t* __restrict__ eln) {
  int row = blockIdx.x * 8 + (threadIdx.x >> 5);
  int lane = threadIdx.x & 31;
  const float4* src = (const float4*)(e + (size_t)row * 128);
  float4 v = src[lane];
  float s = v.x + v.y + v.z + v.w;
#pragma unroll
  for (int o = 16; o; o >>= 1) s += __shfl_xor(s, o, 32);
  float mu = s * (1.0f / 128.0f);
  float dx = v.x - mu, dy = v.y - mu, dz = v.z - mu, dw = v.w - mu;
  float q = dx * dx + dy * dy + dz * dz + dw * dw;
#pragma unroll
  for (int o = 16; o; o >>= 1) q += __shfl_xor(q, o, 32);
  float rs = rsqrtf(q * (1.0f / 128.0f) + 1e-5f);
  int c = lane * 4;
  half_t* dst = eln + (size_t)row * 128 + c;
  dst[0] = (half_t)(dx * rs * g[c + 0] + b[c + 0]);
  dst[1] = (half_t)(dy * rs * g[c + 1] + b[c + 1]);
  dst[2] = (half_t)(dz * rs * g[c + 2] + b[c + 2]);
  dst[3] = (half_t)(dw * rs * g[c + 3] + b[c + 3]);
}

// ---------- QKV projection GEMM -> head-major f16 (Q pre-scaled, bias added) ----------
// qh layout: qh[m*RDIM*128 + row*128 + h*16 + d], m in {Q,K,V}
__global__ void k_qkv(const half_t* __restrict__ eln,
                      const half_t* __restrict__ Wt0, const half_t* __restrict__ Wt1,
                      const float* __restrict__ bq0, const float* __restrict__ bq1,
                      half_t* __restrict__ qh0, half_t* __restrict__ qh1) {
  int pass = blockIdx.y;
  const half_t* Wt = pass ? Wt1 : Wt0;
  const float* bq = pass ? bq1 : bq0;
  half_t* qh = pass ? qh1 : qh0;
  int wave = threadIdx.x >> 5, lane = threadIdx.x & 31;
  int t = blockIdx.x * 8 + wave;          // 55296 tiles: 2304 x 24
  int tc = t % 24, tm = t / 24;
  int r0 = tm * 16, c0 = tc * 16;
  v8f acc = {};
#pragma unroll
  for (int kk = 0; kk < 4; ++kk) {
    v16h a = frag_a(eln, 128, r0, kk * 32, lane);
    v16h b = frag_b(Wt, 128, c0, kk * 32, lane);
    acc = __builtin_amdgcn_wmma_f32_16x16x32_f16(false, a, false, b, (short)0, acc, false, false);
  }
  int n = lane & 15, hi = lane >> 4;
  int c = c0 + n;
  int m = c >> 7;                 // 0=Q 1=K 2=V (tiles never straddle parts)
  int ee = c & 127;
  int d = ee >> 3, h = ee & 7;    // reshape(...,DD,NH): d = e/8, h = e%8
  float sc = (m == 0) ? SCALE : 1.0f;
  float bias = bq[c];
  half_t* outb = qh + (size_t)m * RDIM * 128 + h * 16 + d;
#pragma unroll
  for (int r = 0; r < 8; ++r) {
    int row = r0 + r + hi * 8;
    outb[(size_t)row * 128] = (half_t)((acc[r] + bias) * sc);
  }
}

// ---------- per-head bias slabs: bm[h][i*192+k] = Eproj + mask (outgoing pre-transposed) ----------
__global__ void k_eproj(const half_t* __restrict__ eln,
                        const float* __restrict__ Wei, const float* __restrict__ bei,
                        const float* __restrict__ Weo, const float* __restrict__ beo,
                        const float* __restrict__ mask,
                        float* __restrict__ bmi, float* __restrict__ bmo) {
  int idx = blockIdx.x * 256 + threadIdx.x;   // RDIM*8
  int r = idx >> 3, h = idx & 7;
  const half_t* x = eln + (size_t)r * 128;
  float si = 0.f, so = 0.f;
#pragma unroll 4
  for (int t = 0; t < 128; ++t) {
    float a = (float)x[t];
    si += a * Wei[t * 8 + h];
    so += a * Weo[t * 8 + h];
  }
  float mk = mask[idx];
  bmi[(size_t)h * RDIM + r] = si + bei[h] + mk;                      // E_in[b,i,k,h]+mask[b,i,k,h]
  int i = r % 192, k = r / 192;                                      // r = k*192 + i
  bmo[(size_t)h * RDIM + i * 192 + k] = so + beo[h] + mk;            // E_out[b,k,i,h]+mask[b,k,i,h]
}

// ---------- fused per-(j,h) attention: S = QK^T, softmax(+bias), O = P V ----------
__global__ void k_attn(const half_t* __restrict__ qh0, const half_t* __restrict__ qh1,
                       const float* __restrict__ bmi, const float* __restrict__ bmo,
                       half_t* __restrict__ va) {
  extern __shared__ char smem[];
  half_t* Sh   = (half_t*)smem;                       // [192][SSH] f16 scores -> probs (in place)
  half_t* Qs   = Sh + 192 * SSH;                      // [192][16]
  half_t* Ks   = Qs + 192 * 16;                       // [192][16]
  half_t* Vt   = Ks + 192 * 16;                       // [16][SVT] (V transposed)
  float*  rinv = (float*)(Vt + 16 * SVT);             // [192]

  int pass = blockIdx.y;
  int j = blockIdx.x >> 3;
  int h = blockIdx.x & 7;
  const half_t* qh = pass ? qh1 : qh0;
  const float* bm = (pass ? bmo : bmi) + (size_t)h * RDIM;
  const half_t* Qg = qh;
  const half_t* Kg = qh + (size_t)RDIM * 128;
  const half_t* Vg = Kg + (size_t)RDIM * 128;

  int tid = threadIdx.x;
  if (tid < 192) {
    int qrow = tid * 192 + j;                               // Q[b,i=tid,j]
    int krow = pass ? (tid * 192 + j) : (j * 192 + tid);    // K/V row per pass
    const half_t* qp = Qg + (size_t)qrow * 128 + h * 16;
    const half_t* kp = Kg + (size_t)krow * 128 + h * 16;
    const half_t* vp = Vg + (size_t)krow * 128 + h * 16;
    // Async DMA the 32B Q/K rows straight into LDS (ASYNCcnt path).
    unsigned ldsq = (unsigned)(size_t)(Qs + tid * 16);      // generic->LDS offset (low 32 bits)
    unsigned ldsk = (unsigned)(size_t)(Ks + tid * 16);
    asm volatile("global_load_async_to_lds_b128 %0, %1, off" :: "v"(ldsq), "v"(qp) : "memory");
    asm volatile("global_load_async_to_lds_b128 %0, %1, off offset:16" :: "v"(ldsq), "v"(qp) : "memory");
    asm volatile("global_load_async_to_lds_b128 %0, %1, off" :: "v"(ldsk), "v"(kp) : "memory");
    asm volatile("global_load_async_to_lds_b128 %0, %1, off offset:16" :: "v"(ldsk), "v"(kp) : "memory");
    // V needs a transpose: classic load + LDS scatter
#pragma unroll
    for (int x = 0; x < 16; ++x) Vt[x * SVT + tid] = vp[x];
  }
  asm volatile("s_wait_asynccnt 0" ::: "memory");
  __syncthreads();

  int wave = tid >> 5, lane = tid & 31;
  int n = lane & 15, hi = lane >> 4;

  // --- scores: 12x12 tiles of 16x16, d=16 zero-padded to K=32 ---
  for (int t = wave; t < 144; t += 8) {
    int ti = t / 12, tk = t % 12;
    int i0 = ti * 16, k0 = tk * 16;
    v16h a, b;
    {
      const half_t* p = Qs + (i0 + n) * 16 + hi * 8;
#pragma unroll
      for (int x = 0; x < 8; ++x) a[x] = p[x];
#pragma unroll
      for (int x = 0; x < 8; ++x) a[x + 8] = (half_t)0.f;  // K 16..31 pad
    }
    {
      const half_t* p = Ks + (k0 + n) * 16 + hi * 8;
#pragma unroll
      for (int x = 0; x < 8; ++x) b[x] = p[x];
#pragma unroll
      for (int x = 0; x < 8; ++x) b[x + 8] = (half_t)0.f;  // K 16..31 pad
    }
    v8f acc = {};
    acc = __builtin_amdgcn_wmma_f32_16x16x32_f16(false, a, false, b, (short)0, acc, false, false);
#pragma unroll
    for (int r = 0; r < 8; ++r) {
      int row = i0 + r + hi * 8;
      Sh[row * SSH + k0 + n] = (half_t)acc[r];
    }
  }
  __syncthreads();

  // --- row softmax: bias streamed contiguously from L2, probs written in place, 1/sum deferred ---
  if (tid < 192) {
    half_t* srow = Sh + tid * SSH;
    const float* brow = bm + tid * 192;
    float mx = -1e30f;
    for (int k = 0; k < 192; ++k) {
      float v = (float)srow[k] + brow[k];
      srow[k] = (half_t)v;
      mx = fmaxf(mx, v);
    }
    float sum = 0.f;
    for (int k = 0; k < 192; ++k) {
      float ev = __expf((float)srow[k] - mx);
      srow[k] = (half_t)ev;
      sum += ev;
    }
    rinv[tid] = 1.0f / sum;
  }
  __syncthreads();

  // --- O = P @ V : 12 row tiles, K loop over 6 chunks of 32 ---
  for (int t = wave; t < 12; t += 8) {
    int i0 = t * 16;
    v8f acc = {};
#pragma unroll
    for (int kc = 0; kc < 6; ++kc) {
      v16h a = frag_a(Sh, SSH, i0, kc * 32, lane);
      v16h b = frag_b(Vt, SVT, 0, kc * 32, lane);
      acc = __builtin_amdgcn_wmma_f32_16x16x32_f16(false, a, false, b, (short)0, acc, false, false);
    }
    int d = n;
#pragma unroll
    for (int r = 0; r < 8; ++r) {
      int row = i0 + r + hi * 8;
      float o = acc[r] * rinv[row];
      // concat on head axis: c = d*16 + pass*8 + h
      va[(size_t)(row * 192 + j) * 256 + d * 16 + pass * 8 + h] = (half_t)o;
    }
  }
}

// ---------- output GEMM: out = Va(f16)[R,256] @ Wo + bo (f32 out) ----------
__global__ void k_out(const half_t* __restrict__ va, const half_t* __restrict__ WoT,
                      const float* __restrict__ bo, float* __restrict__ out) {
  int wave = threadIdx.x >> 5, lane = threadIdx.x & 31;
  int r0 = blockIdx.x * 16, c0 = wave * 16;    // 2304 blocks x 8 column tiles
  v8f acc = {};
#pragma unroll
  for (int kk = 0; kk < 8; ++kk) {
    v16h a = frag_a(va, 256, r0, kk * 32, lane);
    v16h b = frag_b(WoT, 256, c0, kk * 32, lane);
    acc = __builtin_amdgcn_wmma_f32_16x16x32_f16(false, a, false, b, (short)0, acc, false, false);
  }
  int n = lane & 15, hi = lane >> 4;
  int c = c0 + n;
  float bias = bo[c];
#pragma unroll
  for (int r = 0; r < 8; ++r) {
    int row = r0 + r + hi * 8;
    out[(size_t)row * 128 + c] = acc[r] + bias;
  }
}

extern "C" void kernel_launch(void* const* d_in, const int* in_sizes, int n_in,
                              void* d_out, int out_size, void* d_ws, size_t ws_size,
                              hipStream_t stream) {
  (void)in_sizes; (void)n_in; (void)out_size; (void)ws_size;
  const float* e    = (const float*)d_in[0];
  const float* mask = (const float*)d_in[1];
  const float* ln_g = (const float*)d_in[2];
  const float* ln_b = (const float*)d_in[3];
  const float* Wq0  = (const float*)d_in[4];
  const float* bq0  = (const float*)d_in[5];
  const float* Wei  = (const float*)d_in[6];
  const float* bei  = (const float*)d_in[7];
  const float* Wq1  = (const float*)d_in[8];
  const float* bq1  = (const float*)d_in[9];
  const float* Weo  = (const float*)d_in[10];
  const float* beo  = (const float*)d_in[11];
  const float* Wo   = (const float*)d_in[12];
  const float* bo   = (const float*)d_in[13];
  float* out = (float*)d_out;

  char* ws = (char*)d_ws;
  auto alloc = [&](size_t bytes) {
    char* p = ws;
    ws += (bytes + 255) & ~(size_t)255;
    return p;
  };
  half_t* eln = (half_t*)alloc((size_t)RDIM * 128 * 2);        //  9.4 MB
  half_t* qh0 = (half_t*)alloc((size_t)3 * RDIM * 128 * 2);    // 28.3 MB
  half_t* qh1 = (half_t*)alloc((size_t)3 * RDIM * 128 * 2);    // 28.3 MB
  half_t* va  = (half_t*)alloc((size_t)RDIM * 256 * 2);        // 18.9 MB
  float*  bmi = (float*)alloc((size_t)8 * RDIM * 4);           //  1.2 MB
  float*  bmo = (float*)alloc((size_t)8 * RDIM * 4);           //  1.2 MB
  half_t* Wt0 = (half_t*)alloc(384 * 128 * 2);
  half_t* Wt1 = (half_t*)alloc(384 * 128 * 2);
  half_t* WoT = (half_t*)alloc(128 * 256 * 2);

  k_prep_w<<<512, 256, 0, stream>>>(Wq0, Wq1, Wo, Wt0, Wt1, WoT);
  k_ln<<<RDIM / 8, 256, 0, stream>>>(e, ln_g, ln_b, eln);
  k_qkv<<<dim3(6912, 2), 256, 0, stream>>>(eln, Wt0, Wt1, bq0, bq1, qh0, qh1);
  k_eproj<<<RDIM * 8 / 256, 256, 0, stream>>>(eln, Wei, bei, Weo, beo, mask, bmi, bmo);
  size_t smem = (size_t)192 * SSH * 2    // score/prob slab (f16)
              + 2 * 192 * 16 * 2         // Qs, Ks
              + 16 * SVT * 2             // Vt
              + 192 * 4;                 // rinv
  k_attn<<<dim3(1536, 2), 256, smem, stream>>>(qh0, qh1, bmi, bmo, va);
  k_out<<<2304, 256, 0, stream>>>(va, WoT, bo, out);
}
